// GCNClassifier_16114717295043
// MI455X (gfx1250) — matrix operand order repaired and checked by reading, not compiled, not run on hardware
//
#include <hip/hip_runtime.h>
#include <stdint.h>

// ---------------------------------------------------------------------------
// GCN classifier for gfx1250 (MI455X): wave32, WMMA bf16 16x16x32,
// pre-packed B fragments staged through LDS, float4-vectorized edge scatter.
// ---------------------------------------------------------------------------

typedef __attribute__((ext_vector_type(16))) __bf16 v16bf;
typedef __attribute__((ext_vector_type(8)))  float  v8f;
typedef unsigned int u32;

#define HDIM 128

static __device__ __forceinline__ __bf16 to_bf16(float x) { return (__bf16)x; }

static __device__ __forceinline__ u32 pack2bf(float lo, float hi) {
  union { __bf16 h[2]; u32 u; } p;
  p.h[0] = (__bf16)lo;
  p.h[1] = (__bf16)hi;
  return p.u;
}

struct BPair { uint4 a, b; };  // 32 bytes == one v16bf fragment

static __device__ __forceinline__ v16bf ld_frag(const uint4* lds, int fragLane) {
  BPair p;
  p.a = lds[fragLane * 2 + 0];
  p.b = lds[fragLane * 2 + 1];
  return __builtin_bit_cast(v16bf, p);
}

// ---------------------------- utility kernels ------------------------------

__global__ void k_zero(float* __restrict__ p, long long n) {
  long long i = (long long)blockIdx.x * blockDim.x + threadIdx.x;
  if (i < n) p[i] = 0.0f;
}

// degrees via f32 atomics
__global__ void k_deg(const int* __restrict__ src, const int* __restrict__ dst,
                      float* __restrict__ din, float* __restrict__ dou, int E) {
  int e = blockIdx.x * blockDim.x + threadIdx.x;
  if (e < E) {
    atomicAdd(&dou[src[e]], 1.0f);
    atomicAdd(&din[dst[e]], 1.0f);
  }
}

// norms + layer-1 scalar feature  s0[i] = in_deg[i] * norm_src[i]
__global__ void k_norm(const float* __restrict__ din, const float* __restrict__ dou,
                       float* __restrict__ nsrc, float* __restrict__ ndst,
                       float* __restrict__ s0, int N) {
  int i = blockIdx.x * blockDim.x + threadIdx.x;
  if (i < N) {
    float di = din[i], dq = dou[i];
    float ns = rsqrtf(fmaxf(dq, 1.0f));
    nsrc[i] = ns;
    ndst[i] = rsqrtf(fmaxf(di, 1.0f));
    s0[i]   = di * ns;
  }
}

// layer-1 edge pass: rank-1 factorization -> scalar scatter only (4B/edge)
__global__ void k_scatter_s(const int* __restrict__ src, const int* __restrict__ dst,
                            const float* __restrict__ s0, float* __restrict__ t, int E) {
  int e = blockIdx.x * blockDim.x + threadIdx.x;
  if (e < E) atomicAdd(&t[dst[e]], s0[src[e]]);
}

// h1[i][j] = relu(t[i]*ndst[i]*W1[j] + b1[j])
__global__ void k_layer1(const float* __restrict__ t, const float* __restrict__ ndst,
                         const float* __restrict__ W1, const float* __restrict__ b1,
                         float* __restrict__ h1, long long total) {
  long long idx = (long long)blockIdx.x * blockDim.x + threadIdx.x;
  if (idx < total) {
    int i = (int)(idx >> 7), j = (int)(idx & (HDIM - 1));
    float v = t[i] * ndst[i] * W1[j] + b1[j];
    h1[idx] = fmaxf(v, 0.0f);
  }
}

// layer-2 edge pass: agg[dst] += h1[src] * nsrc[src].
// One lane handles 4 consecutive columns: float4 gather + 4 f32 atomics.
__global__ void k_scatter_v(const int* __restrict__ src, const int* __restrict__ dst,
                            const float* __restrict__ h1, const float* __restrict__ nsrc,
                            float* __restrict__ agg, long long total /* E*32 */) {
  long long idx = (long long)blockIdx.x * blockDim.x + threadIdx.x;
  if (idx < total) {
    int e = (int)(idx >> 5);
    int q = ((int)idx & 31) * 4;
    int s = src[e], d = dst[e];
    float ns = nsrc[s];
    const float4 v = *(const float4*)(h1 + (long long)s * HDIM + q);
    float* ap = agg + (long long)d * HDIM + q;
    atomicAdd(ap + 0, v.x * ns);
    atomicAdd(ap + 1, v.y * ns);
    atomicAdd(ap + 2, v.z * ns);
    atomicAdd(ap + 3, v.w * ns);
  }
}

// ------------------------ B-fragment pre-packing ---------------------------
// Packs a row-major f32 weight matrix (HDIM x ncols) into the per-lane WMMA
// B-fragment layout (ISA 7.12.2, 16-bit B 32x16): fragment index
// f = (tile*4 + kkidx)*32 + lane; lane half selects K-base {0,8};
// element j -> K = kk + kbase + (j<8?j:8+j), column n = tile*16 + (lane&15).
// Stored as 8 u32 (16 bf16) per fragment-lane. Columns >= ncols are zero.
__global__ void k_packB(const float* __restrict__ W, u32* __restrict__ Bp,
                        int ncols, int total /* frag-lanes = ntiles*128 */) {
  int t = blockIdx.x * blockDim.x + threadIdx.x;
  if (t >= total) return;
  int lane  = t & 31;
  int kkidx = (t >> 5) & 3;
  int tile  = t >> 7;
  int half = lane >> 4, l16 = lane & 15;
  int kbB = half ? 8 : 0;
  int n_b = tile * 16 + l16;
  int kk = kkidx * 32;
  u32* outp = Bp + (long long)t * 8;
#pragma unroll
  for (int jj = 0; jj < 8; ++jj) {
    int k0 = kk + kbB + 2 * jj + ((jj >= 4) ? 8 : 0);
    float lo = (n_b < ncols) ? W[k0 * ncols + n_b]       : 0.0f;
    float hi = (n_b < ncols) ? W[(k0 + 1) * ncols + n_b] : 0.0f;
    outp[jj] = pack2bf(lo, hi);
  }
}

// ------------------- layer-2 GEMM: h2 = relu(agg@W2 * ndst + b2) -----------
// One 16x16 output tile per wave; 8 waves/block cover a 16x128 row strip.
// A fragment (16x32 bf16): lane half K-base {0,8}; elem j -> K=kb+(j<8?j:8+j).
// B fragments come pre-packed (bf16) and are staged via LDS (32 KB/block).
// C/D (16x16 f32): row = r + 8*half, col = lane&15.
__global__ void k_gemm2(const float* __restrict__ agg, const u32* __restrict__ Bp,
                        const float* __restrict__ b2, const float* __restrict__ ndst,
                        float* __restrict__ h2, int N) {
  __shared__ uint4 ldsB[2048];                    // 8 tiles * 4 kk * 32 lanes * 32B
  const int m0 = blockIdx.x * 16;
  if (m0 >= N) return;                            // uniform per block

  // cooperative stage of all packed B fragments (32 KB)
  const uint4* gB = (const uint4*)Bp;
#pragma unroll
  for (int i = threadIdx.x; i < 2048; i += 256) ldsB[i] = gB[i];
  __syncthreads();

  const int wave = threadIdx.x >> 5;
  const int lane = threadIdx.x & 31;
  const int half = lane >> 4;
  const int l16  = lane & 15;
  int m_a = m0 + l16; if (m_a >= N) m_a = N - 1;
  const int kbA = half ? 8 : 0;
  const float* arow = agg + (long long)m_a * HDIM;

  v8f c = {};
#pragma unroll
  for (int kkidx = 0; kkidx < 4; ++kkidx) {
    const int kk = kkidx * 32;
    v16bf a;
#pragma unroll
    for (int j = 0; j < 16; ++j) {
      const int ka = kk + kbA + ((j < 8) ? j : (8 + j));
      a[j] = to_bf16(arow[ka]);
    }
    const v16bf b = ld_frag(ldsB, (wave * 4 + kkidx) * 32 + lane);
    c = __builtin_amdgcn_wmma_f32_16x16x32_bf16(false, a, false, b,
                                                (short)0, c, false, false);
  }

  const int n0 = wave * 16;
#pragma unroll
  for (int r = 0; r < 8; ++r) {
    const int m = m0 + half * 8 + r;
    const int n = n0 + l16;
    if (m < N) {
      float v = c[r] * ndst[m] + b2[n];
      h2[(long long)m * HDIM + n] = fmaxf(v, 0.0f);
    }
  }
}

// ---------------------------- graph pooling --------------------------------
// One lane handles 4 consecutive columns: float4 gather + 4 f32 atomics.
__global__ void k_pool(const int* __restrict__ gid, const float* __restrict__ h2,
                       float* __restrict__ gsum, long long total /* N*32 */) {
  long long idx = (long long)blockIdx.x * blockDim.x + threadIdx.x;
  if (idx < total) {
    int i = (int)(idx >> 5);
    int q = ((int)idx & 31) * 4;
    int g = gid[i];
    const float4 v = *(const float4*)(h2 + (long long)i * HDIM + q);
    float* gp = gsum + (long long)g * HDIM + q;
    atomicAdd(gp + 0, v.x);
    atomicAdd(gp + 1, v.y);
    atomicAdd(gp + 2, v.z);
    atomicAdd(gp + 3, v.w);
  }
}

__global__ void k_cnt(const int* __restrict__ gid, float* __restrict__ gcnt, int N) {
  int i = blockIdx.x * blockDim.x + threadIdx.x;
  if (i < N) atomicAdd(&gcnt[gid[i]], 1.0f);
}

// -------- classifier GEMM: out = (gsum/cnt) @ Wc + bc, C padded to 16 ------
__global__ void k_cls(const float* __restrict__ gsum, const float* __restrict__ gcnt,
                      const u32* __restrict__ Bp, const float* __restrict__ bc,
                      float* __restrict__ out, int G, int C) {
  __shared__ uint4 ldsB[256];                     // 1 tile * 4 kk * 32 lanes * 32B

  // stage first (barrier must be block-uniform; mt varies per wave)
  const uint4* gB = (const uint4*)Bp;
  if (threadIdx.x < 256) ldsB[threadIdx.x] = gB[threadIdx.x];
  __syncthreads();

  const int wave = threadIdx.x >> 5;
  const int lane = threadIdx.x & 31;
  const int mt = blockIdx.x * 8 + wave;
  const int m0 = mt * 16;
  if (m0 >= G) return;

  const int half = lane >> 4;
  const int l16  = lane & 15;
  const int kbA = half ? 8 : 0;
  int m_a = m0 + l16; if (m_a >= G) m_a = G - 1;
  const float inv = 1.0f / fmaxf(gcnt[m_a], 1.0f);
  const float* arow = gsum + (long long)m_a * HDIM;

  v8f c = {};
#pragma unroll
  for (int kkidx = 0; kkidx < 4; ++kkidx) {
    const int kk = kkidx * 32;
    v16bf a;
#pragma unroll
    for (int j = 0; j < 16; ++j) {
      const int ka = kk + kbA + ((j < 8) ? j : (8 + j));
      a[j] = to_bf16(arow[ka] * inv);
    }
    const v16bf b = ld_frag(ldsB, kkidx * 32 + lane);
    c = __builtin_amdgcn_wmma_f32_16x16x32_bf16(false, a, false, b,
                                                (short)0, c, false, false);
  }

#pragma unroll
  for (int r = 0; r < 8; ++r) {
    const int m = m0 + half * 8 + r;
    const int n = l16;
    if (m < G && n < C) out[m * C + n] = c[r] + bc[n];
  }
}

// ------------------------------- launcher ----------------------------------

extern "C" void kernel_launch(void* const* d_in, const int* in_sizes, int n_in,
                              void* d_out, int out_size, void* d_ws, size_t ws_size,
                              hipStream_t stream) {
  (void)n_in; (void)ws_size;
  const int*   src = (const int*)d_in[0];
  const int*   dst = (const int*)d_in[1];
  const int*   gid = (const int*)d_in[2];
  // d_in[3] = num_graphs (device scalar, unused: derived from out_size)
  const float* W1  = (const float*)d_in[4];
  const float* b1  = (const float*)d_in[5];
  const float* W2  = (const float*)d_in[6];
  const float* b2  = (const float*)d_in[7];
  const float* Wc  = (const float*)d_in[8];
  const float* bc  = (const float*)d_in[9];
  float* out = (float*)d_out;

  const int E = in_sizes[0];
  const int N = in_sizes[2];
  const int C = in_sizes[9];          // bc has C elements
  const int G = out_size / C;

  // workspace layout (floats, then u32 pack regions)
  float* ws   = (float*)d_ws;
  float* din  = ws;                               // N
  float* dou  = din  + N;                         // N
  float* nsrc = dou  + N;                         // N
  float* ndst = nsrc + N;                         // N
  float* s0   = ndst + N;                         // N
  float* t    = s0   + N;                         // N
  float* h1   = t    + N;                         // N*H   (reused as h2)
  float* agg  = h1   + (long long)N * HDIM;       // N*H
  float* gsum = agg  + (long long)N * HDIM;       // G*H
  float* gcnt = gsum + (long long)G * HDIM;       // G
  u32*   Bp2  = (u32*)(gcnt + G);                 // 8 tiles*4kk*32 lanes*8 = 8192 u32
  u32*   Bpc  = Bp2 + 8192;                       // 1 tile *4kk*32 lanes*8 = 1024 u32

  const int TB = 256;
  const long long nh = (long long)N * HDIM;

  // zero the accumulator regions (h1 is fully overwritten, skip it)
  {
    long long z1 = 6LL * N;
    k_zero<<<(unsigned)((z1 + TB - 1) / TB), TB, 0, stream>>>(ws, z1);
    long long z2 = nh + (long long)G * HDIM + G;   // agg + gsum + gcnt
    k_zero<<<(unsigned)((z2 + TB - 1) / TB), TB, 0, stream>>>(agg, z2);
  }

  // pack W2 / Wc into WMMA B-fragment bf16 layout (tiny, once per call)
  k_packB<<<4, TB, 0, stream>>>(W2, Bp2, HDIM, 1024);
  k_packB<<<1, 128, 0, stream>>>(Wc, Bpc, C, 128);

  k_deg<<<(unsigned)((E + TB - 1) / TB), TB, 0, stream>>>(src, dst, din, dou, E);
  k_norm<<<(unsigned)((N + TB - 1) / TB), TB, 0, stream>>>(din, dou, nsrc, ndst, s0, N);
  k_scatter_s<<<(unsigned)((E + TB - 1) / TB), TB, 0, stream>>>(src, dst, s0, t, E);
  k_layer1<<<(unsigned)((nh + TB - 1) / TB), TB, 0, stream>>>(t, ndst, W1, b1, h1, nh);

  const long long ev = (long long)E * 32;         // E edges * 32 lanes (4 cols/lane)
  k_scatter_v<<<(unsigned)((ev + TB - 1) / TB), TB, 0, stream>>>(src, dst, h1, nsrc, agg, ev);

  // layer-2 GEMM + bias + norm + relu (WMMA bf16), h2 reuses the h1 buffer
  float* h2 = h1;
  k_gemm2<<<(unsigned)((N + 15) / 16), TB, 0, stream>>>(agg, Bp2, b2, ndst, h2, N);

  const long long pv = (long long)N * 32;         // N nodes * 32 lanes (4 cols/lane)
  k_pool<<<(unsigned)((pv + TB - 1) / TB), TB, 0, stream>>>(gid, h2, gsum, pv);
  k_cnt<<<(unsigned)((N + TB - 1) / TB), TB, 0, stream>>>(gid, gcnt, N);

  // classifier GEMM (WMMA bf16), mean-pool fused into the A-fragment load
  const int mtiles = (G + 15) / 16;
  k_cls<<<(unsigned)((mtiles + 7) / 8), TB, 0, stream>>>(gsum, gcnt, Bpc, bc, out, G, C);
}
